// MLA_56152402428222
// MI455X (gfx1250) — matrix operand order repaired and checked by reading, not compiled, not run on hardware
//
#include <hip/hip_runtime.h>
#include <hip/hip_bf16.h>

typedef __bf16 bf16;
typedef __bf16 v16bf __attribute__((ext_vector_type(16)));
typedef float  v8f   __attribute__((ext_vector_type(8)));
typedef unsigned int u32x4 __attribute__((ext_vector_type(4)));
typedef int          i32x8 __attribute__((ext_vector_type(8)));
typedef int          i32x4 __attribute__((ext_vector_type(4)));

#define B_   32
#define L_   512
#define DM   2048
#define DOWN 512
#define UP   1024
#define NH   16
#define RD   32
#define VD   64
#define QKD  96

__device__ inline v8f wmma_bf16(v16bf a, v16bf b, v8f c) {
  return __builtin_amdgcn_wmma_f32_16x16x32_bf16(false, a, false, b, (short)0, c,
                                                 false, false);
}

__device__ inline unsigned int pack2bf(float a, float b) {
  union { bf16 h[2]; unsigned int u; } x;
  x.h[0] = (bf16)a; x.h[1] = (bf16)b;
  return x.u;
}

// ---------------------------------------------------------------------------
// Tensor Data Mover: 2D bf16 tile load global->LDS (ISA ch.8 D# layout).
//  tile_w elems per row, tile_h rows, row stride in elems, optional LDS pad:
//  pad codes per ISA: interval code v -> 2^(v+1) DWORDs, amount code a -> a+1 DWORDs.
// ---------------------------------------------------------------------------
__device__ inline void tdm_load_2d_bf16(unsigned int lds_addr, const bf16* gptr,
                                        int tile_w, int tile_h, int stride_elems,
                                        int pad_interval_code, int pad_amount_code,
                                        bool pad_enable) {
  unsigned long long ga = (unsigned long long)(uintptr_t)gptr;
  u32x4 g0;
  g0[0] = 1u;                                   // count=1 (valid user descriptor)
  g0[1] = lds_addr;                             // LDS byte address
  g0[2] = (unsigned int)ga;                     // global_addr[31:0]
  g0[3] = (unsigned int)((ga >> 32) & 0x1FFFFFFu) | (2u << 30);  // addr hi | type=2
  i32x8 g1;
  unsigned int pe = pad_enable ? 1u : 0u;
  g1[0] = (int)((1u << 16) | (pe << 20) |
                ((unsigned int)pad_interval_code << 22) |
                ((unsigned int)pad_amount_code << 25));   // data_size=2B, pad cfg
  g1[1] = (int)(((unsigned int)tile_w & 0xFFFFu) << 16);  // tensor_dim0[15:0]
  g1[2] = (int)((((unsigned int)tile_w >> 16) & 0xFFFFu) |
                (((unsigned int)tile_h & 0xFFFFu) << 16)); // dim0 hi | tensor_dim1 lo
  g1[3] = (int)((((unsigned int)tile_h >> 16) & 0xFFFFu) |
                (((unsigned int)tile_w & 0xFFFFu) << 16)); // dim1 hi | tile_dim0
  g1[4] = (int)((unsigned int)tile_h & 0xFFFFu);           // tile_dim1 (tile_dim2=0)
  g1[5] = (int)(unsigned int)stride_elems;                 // tensor_dim0_stride lo32
  g1[6] = 0;
  g1[7] = 0;
  i32x4 z4 = {0, 0, 0, 0};
  i32x8 z8 = {0, 0, 0, 0, 0, 0, 0, 0};
  __builtin_amdgcn_tensor_load_to_lds(g0, g1, z4, z4, z8, 0);
}

__device__ inline unsigned int lds_off32(const void* p) {
  return (unsigned int)(uintptr_t)p;  // LDS aperture: offset = addr[31:0]
}

// ---------------------------------------------------------------------------
// Generic GEMM: C[M,N] = A[M,K] @ W[K,N] + bias   (bf16 WMMA, f32 accumulate)
// Block = 128 threads (4 waves), tile 64x64, K in steps of 32.
// bf16 A tiles are moved by the TDM; fp32 A is converted while staging.
// W tile stored transposed in LDS so B-fragments are contiguous ds_load_b128.
// ---------------------------------------------------------------------------
#define GA_LD  40   // 32 + 8 pad  (pad: 16 dw interval code 3, 4 dw amount code 3)
#define GBT_LD 40

template <typename AT, typename OT>
__global__ __launch_bounds__(128)
void gemm_kernel(const AT* __restrict__ A, const float* __restrict__ W,
                 const float* __restrict__ bias, OT* __restrict__ C,
                 int M, int N, int K) {
  __shared__ bf16 As[64 * GA_LD];
  __shared__ bf16 Bt[64 * GBT_LD];   // transposed: [n][k]
  const int tid  = threadIdx.x;
  const int lane = tid & 31;
  const int wave = tid >> 5;
  const int hgrp = lane >> 4;
  const int ln   = lane & 15;
  const int mbase = blockIdx.y * 64;
  const int nbase = blockIdx.x * 64;
  const unsigned int as_base = lds_off32(As);

  v8f acc[4] = {};

  for (int kk = 0; kk < K; kk += 32) {
    if constexpr (sizeof(AT) == 2) {
      // A tile 64x32 bf16 via Tensor Data Mover (pads LDS rows to GA_LD)
      if (wave == 0)
        tdm_load_2d_bf16(as_base, (const bf16*)&A[(size_t)mbase * K + kk],
                         32, 64, K, 3, 3, true);
    } else {
      // A tile 64x32 fp32 -> bf16 (vectorized)
#pragma unroll
      for (int i = 0; i < 4; ++i) {
        int idx = tid + i * 128;          // 512 float4 chunks
        int r = idx >> 3, c4 = (idx & 7) * 4;
        float4 a = *(const float4*)&A[(size_t)(mbase + r) * K + kk + c4];
        *(unsigned int*)&As[r * GA_LD + c4]     = pack2bf(a.x, a.y);
        *(unsigned int*)&As[r * GA_LD + c4 + 2] = pack2bf(a.z, a.w);
      }
    }
    // W tile 32x64 fp32 -> bf16, transposed into Bt[n][k]
#pragma unroll
    for (int i = 0; i < 4; ++i) {
      int idx = tid + i * 128;            // 512 float4 chunks
      int r = idx >> 4, c4 = (idx & 15) * 4;
      int n = nbase + c4;
      float4 w = {0.0f, 0.0f, 0.0f, 0.0f};
      if (n < N) w = *(const float4*)&W[(size_t)(kk + r) * N + n];
      Bt[(c4 + 0) * GBT_LD + r] = (bf16)w.x;
      Bt[(c4 + 1) * GBT_LD + r] = (bf16)w.y;
      Bt[(c4 + 2) * GBT_LD + r] = (bf16)w.z;
      Bt[(c4 + 3) * GBT_LD + r] = (bf16)w.w;
    }
    if constexpr (sizeof(AT) == 2) {
      if (wave == 0) __builtin_amdgcn_s_wait_tensorcnt(0);
    }
    __syncthreads();

    // A fragment: lane half h holds k = 16h + e (contiguous 16-elem run)
    const bf16* Ab = &As[wave * 16 * GA_LD];
    v16bf afrag;
#pragma unroll
    for (int e = 0; e < 16; ++e) {
      int v = e >> 1;
      int k = hgrp * 16 + e;
      afrag[e] = Ab[ln * GA_LD + k];
    }
#pragma unroll
    for (int t = 0; t < 4; ++t) {
      v16bf bvec;   // B frag: lane = N, halves split K; contiguous in e
#pragma unroll
      for (int e = 0; e < 16; ++e)
        bvec[e] = Bt[(t * 16 + ln) * GBT_LD + hgrp * 16 + e];
      acc[t] = wmma_bf16(afrag, bvec, acc[t]);
    }
    __syncthreads();
  }

#pragma unroll
  for (int t = 0; t < 4; ++t) {
#pragma unroll
    for (int i = 0; i < 8; ++i) {
      int row = mbase + wave * 16 + hgrp * 8 + i;
      int col = nbase + t * 16 + ln;
      if (col < N) C[(size_t)row * N + col] = (OT)(acc[t][i] + bias[col]);
    }
  }
}

// ---------------------------------------------------------------------------
// RoPE (faithful to source quirk: 'cos' cache holds sin, 'sin' cache holds cos)
// ---------------------------------------------------------------------------
__global__ void rope_kernel(const bf16* __restrict__ in, bf16* __restrict__ out,
                            int tokens, int heads) {
  int total = tokens * heads * (RD / 2);
  for (int p = blockIdx.x * blockDim.x + threadIdx.x; p < total;
       p += gridDim.x * blockDim.x) {
    int j    = p % (RD / 2);
    int rest = p / (RD / 2);
    int hh   = rest % heads;
    int tok  = rest / heads;
    int l    = tok % L_;
    float theta = __powf(10000.0f, -(float)j / 16.0f);
    float ang = (float)l * theta;
    float s, c;
    __sincosf(ang, &s, &c);
    size_t base = (size_t)tok * heads * RD + (size_t)hh * RD + 2 * j;
    float x0 = (float)in[base];
    float x1 = (float)in[base + 1];
    out[base]     = (bf16)(x0 * s - x1 * c);
    out[base + 1] = (bf16)(x1 * s + x0 * c);
  }
}

// ---------------------------------------------------------------------------
// Flash attention: 4 waves/block, wave owns 16 q rows, keys in blocks of 32.
// K/V tile staging via TDM (k_c & v_c per-head 64-wide slices, stride 1024);
// rope part of K staged manually into the padded columns.
// ---------------------------------------------------------------------------
#define QK_LD 104   // 64 + 40 pad  (interval code 4 = 32 dw, amount code 19 = 20 dw)
#define VV_LD 72    // 64 + 8 pad   (interval code 4,        amount code 3  = 4 dw)
#define P_LD  36

__global__ __launch_bounds__(128)
void attn_kernel(const bf16* __restrict__ qc, const bf16* __restrict__ qr,
                 const bf16* __restrict__ kc, const bf16* __restrict__ kr,
                 const bf16* __restrict__ vc, bf16* __restrict__ ao) {
  __shared__ bf16 q_lds[64 * QK_LD];
  __shared__ bf16 k_lds[32 * QK_LD];
  __shared__ bf16 v_lds[32 * VV_LD];
  __shared__ bf16 p_lds[4 * 16 * P_LD];

  const int tid  = threadIdx.x;
  const int lane = tid & 31;
  const int wave = tid >> 5;
  const int hgrp = lane >> 4;
  const int ln   = lane & 15;

  const int bh   = blockIdx.x >> 3;
  const int qblk = blockIdx.x & 7;
  const int b    = bh >> 4;
  const int h    = bh & 15;
  const float scale = 0.103568666f;  // 1/(sqrt(16)+sqrt(32))
  const unsigned int klds_base = lds_off32(k_lds);
  const unsigned int vlds_base = lds_off32(v_lds);

  // stage this wave's 16 q rows: [16 x 96] = qc(64) ++ roped qr(32)
  const int qrow0 = qblk * 64 + wave * 16;
  bf16* qw = &q_lds[wave * 16 * QK_LD];
  for (int idx = lane; idx < 16 * QKD; idx += 32) {
    int r = idx / QKD, c = idx % QKD;
    size_t tok = (size_t)b * L_ + qrow0 + r;
    qw[r * QK_LD + c] = (c < VD) ? qc[tok * UP + h * VD + c]
                                 : qr[tok * (NH * RD) + h * RD + (c - VD)];
  }
  asm volatile("s_wait_dscnt 0" ::: "memory");

  v16bf qf[3];
#pragma unroll
  for (int d = 0; d < 3; ++d) {
#pragma unroll
    for (int e = 0; e < 16; ++e) {
      int v = e >> 1;
      int k = hgrp * 16 + e;
      qf[d][e] = qw[ln * QK_LD + d * 32 + k];
    }
  }

  v8f acc[4] = {};
  float mrow[8], lrow[8];
#pragma unroll
  for (int i = 0; i < 8; ++i) { mrow[i] = -1e30f; lrow[i] = 0.0f; }

  for (int kb = 0; kb < L_; kb += 32) {
    __syncthreads();   // previous iteration's LDS consumers done
    if (wave == 0) {
      // k_c slice: 32 rows x 64 elems, row stride 1024; pad rows to QK_LD
      tdm_load_2d_bf16(klds_base,
                       &kc[((size_t)b * L_ + kb) * UP + (size_t)h * VD],
                       64, 32, UP, 4, 19, true);
      // v_c slice: 32 rows x 64 elems; pad rows to VV_LD
      tdm_load_2d_bf16(vlds_base,
                       &vc[((size_t)b * L_ + kb) * UP + (size_t)h * VD],
                       64, 32, UP, 4, 3, true);
    }
    // rope columns of K (broadcast across heads) staged manually
    for (int idx = tid; idx < 32 * RD; idx += 128) {
      int r = idx >> 5, c = idx & 31;
      k_lds[r * QK_LD + VD + c] = kr[((size_t)b * L_ + kb + r) * RD + c];
    }
    if (wave == 0) __builtin_amdgcn_s_wait_tensorcnt(0);
    __syncthreads();

    // scores: S (16q x 32k) as two 16x16 tiles, 3 WMMAs each over d=96
    v8f S[2] = {};
#pragma unroll
    for (int nc = 0; nc < 2; ++nc) {
#pragma unroll
      for (int d = 0; d < 3; ++d) {
        v16bf bvec;  // B = K^T : (K=dim, N=key); contiguous in e
#pragma unroll
        for (int e = 0; e < 16; ++e)
          bvec[e] = k_lds[(nc * 16 + ln) * QK_LD + d * 32 + hgrp * 16 + e];
        S[nc] = wmma_bf16(qf[d], bvec, S[nc]);
      }
    }

    // online softmax per row (C layout: row m = hgrp*8+i across 16 lanes)
    bf16* pw = &p_lds[wave * 16 * P_LD];
#pragma unroll
    for (int i = 0; i < 8; ++i) {
      float s0 = S[0][i] * scale;
      float s1 = S[1][i] * scale;
      float r = fmaxf(s0, s1);
#pragma unroll
      for (int off = 1; off < 16; off <<= 1)
        r = fmaxf(r, __shfl_xor(r, off, 32));
      float mnew = fmaxf(mrow[i], r);
      float corr = __expf(mrow[i] - mnew);
      float p0 = __expf(s0 - mnew);
      float p1 = __expf(s1 - mnew);
      float rs = p0 + p1;
#pragma unroll
      for (int off = 1; off < 16; off <<= 1)
        rs += __shfl_xor(rs, off, 32);
      lrow[i] = lrow[i] * corr + rs;
      mrow[i] = mnew;
#pragma unroll
      for (int t = 0; t < 4; ++t) acc[t][i] *= corr;
      pw[(hgrp * 8 + i) * P_LD + ln]      = (bf16)p0;
      pw[(hgrp * 8 + i) * P_LD + 16 + ln] = (bf16)p1;
    }
    asm volatile("s_wait_dscnt 0" ::: "memory");

    // P as A-fragment (transpose via LDS), then P @ V (4 WMMAs over vdim)
    v16bf pf;
#pragma unroll
    for (int e = 0; e < 16; ++e) {
      int v = e >> 1;
      int k = hgrp * 16 + e;
      pf[e] = pw[ln * P_LD + k];
    }
#pragma unroll
    for (int t = 0; t < 4; ++t) {
      v16bf vvec;
#pragma unroll
      for (int e = 0; e < 16; ++e)
        vvec[e] = v_lds[(hgrp * 16 + e) * VV_LD + t * 16 + ln];
      acc[t] = wmma_bf16(pf, vvec, acc[t]);
    }
  }

  // normalize + store as [token, h*64+vd] bf16 (ready for final GEMM)
#pragma unroll
  for (int t = 0; t < 4; ++t) {
#pragma unroll
    for (int i = 0; i < 8; ++i) {
      int row = qrow0 + hgrp * 8 + i;
      size_t tok = (size_t)b * L_ + row;
      ao[tok * UP + h * VD + t * 16 + ln] = (bf16)(acc[t][i] / lrow[i]);
    }
  }
}

// ---------------------------------------------------------------------------
extern "C" void kernel_launch(void* const* d_in, const int* in_sizes, int n_in,
                              void* d_out, int out_size, void* d_ws, size_t ws_size,
                              hipStream_t stream) {
  const float* h     = (const float*)d_in[0];
  const float* W_dkv = (const float*)d_in[1];
  const float* b_dkv = (const float*)d_in[2];
  const float* W_uk  = (const float*)d_in[3];
  const float* b_uk  = (const float*)d_in[4];
  const float* W_uv  = (const float*)d_in[5];
  const float* b_uv  = (const float*)d_in[6];
  const float* W_dq  = (const float*)d_in[7];
  const float* b_dq  = (const float*)d_in[8];
  const float* W_uq  = (const float*)d_in[9];
  const float* b_uq  = (const float*)d_in[10];
  const float* W_qr  = (const float*)d_in[11];
  const float* b_qr  = (const float*)d_in[12];
  const float* W_kr  = (const float*)d_in[13];
  const float* b_kr  = (const float*)d_in[14];
  const float* W_fc  = (const float*)d_in[15];
  const float* b_fc  = (const float*)d_in[16];
  float* out = (float*)d_out;

  char* ws = (char*)d_ws;
  const size_t MB = 1024 * 1024;
  bf16* cq  = (bf16*)(ws + 0);        // [16384, 512]
  bf16* ckv = (bf16*)(ws + 16  * MB); // [16384, 512]
  bf16* qcb = (bf16*)(ws + 32  * MB); // [16384, 1024]
  bf16* kcb = (bf16*)(ws + 64  * MB); // [16384, 1024]
  bf16* vcb = (bf16*)(ws + 96  * MB); // [16384, 1024]
  bf16* qrb = (bf16*)(ws + 128 * MB); // [16384, 512]
  bf16* qr2 = (bf16*)(ws + 144 * MB); // [16384, 512] roped
  bf16* krb = (bf16*)(ws + 160 * MB); // [16384, 32]
  bf16* kr2 = (bf16*)(ws + 161 * MB); // [16384, 32] roped
  bf16* aob = (bf16*)(ws + 162 * MB); // [16384, 1024]

  const int M = B_ * L_;  // 16384 tokens
  dim3 blk(128);

  gemm_kernel<float, bf16><<<dim3(DOWN / 64, M / 64), blk, 0, stream>>>(
      h, W_dkv, b_dkv, ckv, M, DOWN, DM);
  gemm_kernel<float, bf16><<<dim3(DOWN / 64, M / 64), blk, 0, stream>>>(
      h, W_dq, b_dq, cq, M, DOWN, DM);
  gemm_kernel<float, bf16><<<dim3(1, M / 64), blk, 0, stream>>>(
      h, W_kr, b_kr, krb, M, RD, DM);

  gemm_kernel<bf16, bf16><<<dim3(UP / 64, M / 64), blk, 0, stream>>>(
      ckv, W_uk, b_uk, kcb, M, UP, DOWN);
  gemm_kernel<bf16, bf16><<<dim3(UP / 64, M / 64), blk, 0, stream>>>(
      ckv, W_uv, b_uv, vcb, M, UP, DOWN);
  gemm_kernel<bf16, bf16><<<dim3(UP / 64, M / 64), blk, 0, stream>>>(
      cq, W_uq, b_uq, qcb, M, UP, DOWN);
  gemm_kernel<bf16, bf16><<<dim3((NH * RD) / 64, M / 64), blk, 0, stream>>>(
      cq, W_qr, b_qr, qrb, M, NH * RD, DOWN);

  rope_kernel<<<512, 256, 0, stream>>>(qrb, qr2, M, NH);
  rope_kernel<<<64, 256, 0, stream>>>(krb, kr2, M, 1);

  attn_kernel<<<B_ * NH * 8, 128, 0, stream>>>(qcb, qr2, kcb, kr2, vcb, aob);

  gemm_kernel<bf16, float><<<dim3(DM / 64, M / 64), blk, 0, stream>>>(
      aob, W_fc, b_fc, out, M, DM, UP);
}
